// EnhancedGNNEncoder_25400436589082
// MI455X (gfx1250) — matrix operand order repaired and checked by reading, not compiled, not run on hardware
//
#include <hip/hip_runtime.h>
#include <hip/hip_bf16.h>
#include <math.h>

// ---------------------------------------------------------------------------
// Types for CDNA5 WMMA (wave32): v_wmma_f32_16x16x32_bf16
// ---------------------------------------------------------------------------
typedef __attribute__((ext_vector_type(16))) __bf16 v16bf;
typedef __attribute__((ext_vector_type(8)))  __bf16 v8bf;
typedef __attribute__((ext_vector_type(8)))  float  v8f;

#define DD 64

__device__ __forceinline__ __bf16 to_bf16(float x)  { return (__bf16)x; }
__device__ __forceinline__ __bf16 to_bf16(__bf16 x) { return x; }
// fast sigmoid: v_exp_f32 + v_rcp_f32 (no IEEE div chain)
__device__ __forceinline__ float sigmoidf_(float x) {
  return __builtin_amdgcn_rcpf(1.0f + __expf(-x));
}

// ---------------------------------------------------------------------------
// zero fill
// ---------------------------------------------------------------------------
__global__ void zero_kernel(float* __restrict__ p, long n) {
  long i = (long)blockIdx.x * blockDim.x + threadIdx.x;
  if (i < n) p[i] = 0.0f;
}

// ---------------------------------------------------------------------------
// Transpose W[NL,K,64] (f32, row-major) -> Wt[NL,64,K] (bf16) so that each
// WMMA B fragment is read as two contiguous 8-element runs along K.
// ---------------------------------------------------------------------------
__global__ void transpose_w_kernel(const float* __restrict__ W, __bf16* __restrict__ Wt,
                                   int K, long total) {
  long gid = (long)blockIdx.x * blockDim.x + threadIdx.x;
  if (gid >= total) return;
  long per = (long)K * DD;
  int l   = (int)(gid / per);
  int rem = (int)(gid - (long)l * per);
  int k = rem / DD, n = rem % DD;
  Wt[((long)l * DD + n) * K + k] = (__bf16)W[gid];
}

// ---------------------------------------------------------------------------
// Embedding MLP: out = relu(x@W1+b1)@W2+b2, DIN small (5/19/30). VALU kernel:
// 64 threads per node, hidden staged in LDS. One-shot cost, not worth WMMA.
// ---------------------------------------------------------------------------
template <int DIN>
__global__ __launch_bounds__(256) void embed_kernel(
    const float* __restrict__ x, const float* __restrict__ w1, const float* __restrict__ b1,
    const float* __restrict__ w2, const float* __restrict__ b2,
    float* __restrict__ out, int N) {
  __shared__ float hid_s[4][DD];
  int nl = threadIdx.x >> 6;       // node within block (4 per block)
  int j  = threadIdx.x & 63;
  int n  = blockIdx.x * 4 + nl;
  float* hid = hid_s[nl];
  if (n < N) {
    float h = b1[j];
    const float* xr = x + (long)n * DIN;
#pragma unroll
    for (int k = 0; k < DIN; ++k) h += xr[k] * w1[k * DD + j];
    hid[j] = fmaxf(h, 0.0f);
  }
  __syncthreads();
  if (n < N) {
    float o = b2[j];
#pragma unroll
    for (int k = 0; k < DD; ++k) o += hid[k] * w2[k * DD + j];
    out[(long)n * DD + j] = o;
  }
}

// ---------------------------------------------------------------------------
// Edge weight module: sigmoid(relu([h_src,h_dst,ef]@W1+b1)@W2+b2), K=136.
// One wave32 per edge; comb staged in LDS; final dot via wave shfl reduction.
// ---------------------------------------------------------------------------
__global__ __launch_bounds__(128) void edge_weight_kernel(
    const float* __restrict__ hsrc, const float* __restrict__ hdst,
    const int* __restrict__ src, const int* __restrict__ dst,
    const float* __restrict__ ef,
    const float* __restrict__ w1, const float* __restrict__ b1,
    const float* __restrict__ w2, const float* __restrict__ b2,
    float* __restrict__ out, long E) {
  __shared__ float comb_s[4][136];
  int wave = threadIdx.x >> 5, lane = threadIdx.x & 31;
  long e = (long)blockIdx.x * 4 + wave;
  float* comb = comb_s[wave];
  if (e < E) {
    int s = src[e], t = dst[e];
    for (int i = lane; i < DD; i += 32) comb[i]      = hsrc[(long)s * DD + i];
    for (int i = lane; i < DD; i += 32) comb[64 + i] = hdst[(long)t * DD + i];
    if (lane < 8) comb[128 + lane] = ef[e * 8 + lane];
  }
  __syncthreads();
  if (e < E) {
    float acc = 0.0f;
#pragma unroll
    for (int jj = 0; jj < 2; ++jj) {
      int j = lane + jj * 32;
      float h = b1[j];
      for (int k = 0; k < 136; ++k) h += comb[k] * w1[k * DD + j];
      acc += fmaxf(h, 0.0f) * w2[j];
    }
#pragma unroll
    for (int off = 16; off > 0; off >>= 1) acc += __shfl_xor(acc, off, 32);
    if (lane == 0) out[e] = sigmoidf_(acc + b2[0]);
  }
}

// ---------------------------------------------------------------------------
// WMMA GEMM: C[N,64] = act(A[N,K] @ W[K,64] + bias), K in {64,128}.
// 128 threads = 4 waves; block covers 128 rows; each wave owns 2 row tiles
// (16 rows each) so every B fragment feeds 2 WMMAs. A and Wt staged in LDS
// (bf16). Paddings chosen so fragment reads are 16B-aligned AND conflict-free:
//   As stride = K+8  elems -> 36/68 dwords, gcd(·,64)=4 -> 16 disjoint spans
//   Ws stride = K+24 elems -> 44/76 dwords, gcd(·,64)=4 -> 16 disjoint spans
// Lane layouts per CDNA5 ISA §7.12.2:
//   A 16x32 : lane<16 -> M=lane,   K {ks..+7, ks+16..+23}
//             lane>=16-> M=lane-16,K {ks+8..+15, ks+24..+31}
//   B 32x16 : N=lane&15, K {ks+(lane<16?0:8)..+7, ks+16+(lane<16?0:8)..+7}
//   C 16x16 : N=lane&15, VGPR r -> M = r + (lane<16?0:8)
// ACT: 0=linear 1=relu 2=sigmoid (compile-time -> branchless epilogue)
// ---------------------------------------------------------------------------
template <int K, typename AT, int ACT>
__global__ __launch_bounds__(128) void gemm_n64_kernel(
    const AT* __restrict__ A, const __bf16* __restrict__ Wt,
    const float* __restrict__ bias, float* __restrict__ C, int N) {
  constexpr int BR = 128;  // rows per block
  __shared__ alignas(16) __bf16 As[BR][K + 8];
  __shared__ alignas(16) __bf16 Ws[64][K + 24];
  const int tid = threadIdx.x;
  const int rb  = blockIdx.x * BR;

  for (int idx = tid; idx < BR * K; idx += 128) {
    int r = idx / K, c = idx % K;
    int row = rb + r;
    AT v = (row < N) ? A[(long)row * K + c] : (AT)0.0f;
    As[r][c] = to_bf16(v);
  }
  for (int idx = tid; idx < 64 * K; idx += 128) {
    int n = idx / K, k = idx % K;
    Ws[n][k] = Wt[(long)n * K + k];
  }
  __syncthreads();

  const int wave  = tid >> 5;
  const int lane  = tid & 31;
  const int lhalf = (lane >> 4) & 1;
  const int l15   = lane & 15;

  v8f cacc[2][4] = {};

#pragma unroll
  for (int ks = 0; ks < K; ks += 32) {
    // ---- A fragments for this wave's two row tiles ----
    v16bf a[2];
#pragma unroll
    for (int rt = 0; rt < 2; ++rt) {
      const int am = (wave * 2 + rt) * 16 + l15;
      const int ak = ks + lhalf * 8;
      v8bf alo = *(const v8bf*)&As[am][ak];
      v8bf ahi = *(const v8bf*)&As[am][ak + 16];
#pragma unroll
      for (int i = 0; i < 8; ++i) { a[rt][i] = alo[i]; a[rt][i + 8] = ahi[i]; }
    }

    const int bk = ks + lhalf * 8;
#pragma unroll
    for (int t = 0; t < 4; ++t) {
      const int bn = t * 16 + l15;
      v8bf blo = *(const v8bf*)&Ws[bn][bk];
      v8bf bhi = *(const v8bf*)&Ws[bn][bk + 16];
      v16bf b;
#pragma unroll
      for (int i = 0; i < 8; ++i) { b[i] = blo[i]; b[i + 8] = bhi[i]; }
#pragma unroll
      for (int rt = 0; rt < 2; ++rt) {
        cacc[rt][t] = __builtin_amdgcn_wmma_f32_16x16x32_bf16(
            false, a[rt], false, b, (short)0, cacc[rt][t], false, false);
      }
    }
  }

  float bv[4];
#pragma unroll
  for (int t = 0; t < 4; ++t) bv[t] = bias[t * 16 + l15];

#pragma unroll
  for (int rt = 0; rt < 2; ++rt) {
#pragma unroll
    for (int r = 0; r < 8; ++r) {
      int M = r + lhalf * 8;
      int row = rb + (wave * 2 + rt) * 16 + M;
      if (row < N) {
#pragma unroll
        for (int t = 0; t < 4; ++t) {
          float v = cacc[rt][t][r] + bv[t];
          if (ACT == 1) v = fmaxf(v, 0.0f);
          else if (ACT == 2) v = sigmoidf_(v);
          C[(long)row * DD + t * 16 + l15] = v;
        }
      }
    }
  }
}

// ---------------------------------------------------------------------------
// Edge scatter: agg[dst] += msg[src]*w, deg[dst] += w. Thread per (edge,dim);
// fp32 atomics land in L2 (agg+msg working set < 192MB L2). Edge-list
// look-ahead via global_prefetch.
// ---------------------------------------------------------------------------
__global__ void scatter_kernel(const float* __restrict__ msg, const int* __restrict__ src,
                               const int* __restrict__ dst, const float* __restrict__ ew,
                               float* __restrict__ agg, float* __restrict__ deg, long E) {
  long gid = (long)blockIdx.x * blockDim.x + threadIdx.x;
  if (gid >= E * DD) return;
  long e = gid >> 6;
  int  d = (int)(gid & 63);
  if (d == 0) {
    long e2 = e + 4096;
    if (e2 < E) {
      __builtin_prefetch(&src[e2], 0, 0);
      __builtin_prefetch(&dst[e2], 0, 0);
      __builtin_prefetch(&ew[e2], 0, 0);
    }
  }
  float w = ew[e];
  int s = src[e], t = dst[e];
  atomicAdd(&agg[(long)t * DD + d], msg[(long)s * DD + d] * w);
  if (d == 0) atomicAdd(&deg[t], w);
}

// ---------------------------------------------------------------------------
// comb = [agg/max(deg,1), node] stored directly as bf16 (A operand for the
// K=128 gate/update GEMMs -> no re-conversion in the GEMM stage-in).
// ---------------------------------------------------------------------------
__global__ void build_comb_kernel(const float* __restrict__ agg, const float* __restrict__ deg,
                                  const float* __restrict__ node, __bf16* __restrict__ comb,
                                  int N) {
  long gid = (long)blockIdx.x * blockDim.x + threadIdx.x;
  if (gid >= (long)N * DD) return;
  int n = (int)(gid >> 6), d = (int)(gid & 63);
  float dg = fmaxf(deg[n], 1.0f);
  comb[(long)n * 128 + d]      = (__bf16)(agg[gid] / dg);
  comb[(long)n * 128 + 64 + d] = (__bf16)node[gid];
}

// ---------------------------------------------------------------------------
// out = LayerNorm(gate*upd + (1-gate)*node), eps=1e-3. One wave32 per row,
// 2 elems/lane, shfl_xor tree reductions (wave32-native). In-place on node.
// ---------------------------------------------------------------------------
__global__ __launch_bounds__(128) void finalize_ln_kernel(
    const float* __restrict__ gate, const float* __restrict__ upd,
    const float* __restrict__ lng, const float* __restrict__ lnb,
    float* __restrict__ node, int N) {
  int row  = (blockIdx.x * blockDim.x + threadIdx.x) >> 5;
  int lane = threadIdx.x & 31;
  if (row >= N) return;
  long base = (long)row * DD;
  float x[2];
#pragma unroll
  for (int h = 0; h < 2; ++h) {
    int j = lane + 32 * h;
    float g = gate[base + j];
    x[h] = g * upd[base + j] + (1.0f - g) * node[base + j];
  }
  float s = x[0] + x[1], s2 = x[0] * x[0] + x[1] * x[1];
#pragma unroll
  for (int off = 16; off > 0; off >>= 1) {
    s  += __shfl_xor(s, off, 32);
    s2 += __shfl_xor(s2, off, 32);
  }
  float mean = s * (1.0f / 64.0f);
  float var  = s2 * (1.0f / 64.0f) - mean * mean;
  float inv  = rsqrtf(var + 1e-3f);
#pragma unroll
  for (int h = 0; h < 2; ++h) {
    int j = lane + 32 * h;
    node[base + j] = (x[h] - mean) * inv * lng[j] + lnb[j];
  }
}

// ---------------------------------------------------------------------------
// Host orchestration
// ---------------------------------------------------------------------------
static inline unsigned cdivu(long a, long b) { return (unsigned)((a + b - 1) / b); }

extern "C" void kernel_launch(void* const* d_in, const int* in_sizes, int n_in,
                              void* d_out, int out_size, void* d_ws, size_t ws_size,
                              hipStream_t stream) {
  (void)in_sizes; (void)n_in; (void)out_size; (void)ws_size;
  const int  NVAR = 100000, NCONS = 50000, NCUT = 20000, NL = 16;
  const long EVC = 800000, EVK = 200000;

  // ---- inputs (setup_inputs dict order) ----
  const float* var_feat  = (const float*)d_in[0];
  const float* cons_feat = (const float*)d_in[1];
  const float* cut_feat  = (const float*)d_in[2];
  const float* vc_ef     = (const float*)d_in[3];
  const float* vk_ef     = (const float*)d_in[4];
  const int*   vc_edges  = (const int*)d_in[5];
  const int*   vk_edges  = (const int*)d_in[6];
  const int *vc_s = vc_edges, *vc_d = vc_edges + EVC;
  const int *vk_s = vk_edges, *vk_d = vk_edges + EVK;

  const float *var_w1 = (const float*)d_in[7],  *var_b1 = (const float*)d_in[8];
  const float *var_w2 = (const float*)d_in[9],  *var_b2 = (const float*)d_in[10];
  const float *cons_w1 = (const float*)d_in[11], *cons_b1 = (const float*)d_in[12];
  const float *cons_w2 = (const float*)d_in[13], *cons_b2 = (const float*)d_in[14];
  const float *cut_w1 = (const float*)d_in[15], *cut_b1 = (const float*)d_in[16];
  const float *cut_w2 = (const float*)d_in[17], *cut_b2 = (const float*)d_in[18];
  const float *ewvc_w1 = (const float*)d_in[19], *ewvc_b1 = (const float*)d_in[20];
  const float *ewvc_w2 = (const float*)d_in[21], *ewvc_b2 = (const float*)d_in[22];
  const float *ewvk_w1 = (const float*)d_in[23], *ewvk_b1 = (const float*)d_in[24];
  const float *ewvk_w2 = (const float*)d_in[25], *ewvk_b2 = (const float*)d_in[26];
  const float *mp_msg_w1 = (const float*)d_in[27], *mp_msg_b1 = (const float*)d_in[28];
  const float *mp_msg_w2 = (const float*)d_in[29], *mp_msg_b2 = (const float*)d_in[30];
  const float *mp_gate_w = (const float*)d_in[31], *mp_gate_b = (const float*)d_in[32];
  const float *mp_upd_w1 = (const float*)d_in[33], *mp_upd_b1 = (const float*)d_in[34];
  const float *mp_upd_w2 = (const float*)d_in[35], *mp_upd_b2 = (const float*)d_in[36];
  const float *mp_ln_g = (const float*)d_in[37], *mp_ln_b = (const float*)d_in[38];

  // ---- workspace carve (256B aligned) ----
  char*  ws  = (char*)d_ws;
  size_t off = 0;
  auto carve = [&](size_t bytes) -> void* {
    void* p = ws + off;
    off += (bytes + 255) & ~(size_t)255;
    return p;
  };
  float*  h_var  = (float*)carve((size_t)NVAR * DD * 4);
  float*  h_cons = (float*)carve((size_t)NCONS * DD * 4);
  float*  h_cut  = (float*)d_out;  // output buffer doubles as h_cut (20000x64 f32)
  float*  vcw  = (float*)carve((size_t)EVC * 4);
  float*  vkw  = (float*)carve((size_t)EVK * 4);
  float*  bufA = (float*)carve((size_t)NVAR * DD * 4);   // msg hidden / upd hidden
  float*  bufB = (float*)carve((size_t)NVAR * DD * 4);   // msg, then gate
  float*  bufD = (float*)carve((size_t)NVAR * DD * 4);   // upd
  float*  agg  = (float*)carve((size_t)NVAR * DD * 4);
  float*  deg  = (float*)carve((size_t)NVAR * 4);
  __bf16* comb = (__bf16*)carve((size_t)NVAR * 128 * 2);
  __bf16* msgw1t = (__bf16*)carve((size_t)NL * DD * DD * 2);
  __bf16* msgw2t = (__bf16*)carve((size_t)NL * DD * DD * 2);
  __bf16* gatewt = (__bf16*)carve((size_t)NL * DD * 128 * 2);
  __bf16* updw1t = (__bf16*)carve((size_t)NL * DD * 128 * 2);
  __bf16* updw2t = (__bf16*)carve((size_t)NL * DD * DD * 2);

  // ---- pre-transpose MP weights to bf16 column-major ----
  {
    long t64  = (long)NL * 64 * DD;
    long t128 = (long)NL * 128 * DD;
    transpose_w_kernel<<<cdivu(t64, 256), 256, 0, stream>>>(mp_msg_w1, msgw1t, 64, t64);
    transpose_w_kernel<<<cdivu(t64, 256), 256, 0, stream>>>(mp_msg_w2, msgw2t, 64, t64);
    transpose_w_kernel<<<cdivu(t128, 256), 256, 0, stream>>>(mp_gate_w, gatewt, 128, t128);
    transpose_w_kernel<<<cdivu(t128, 256), 256, 0, stream>>>(mp_upd_w1, updw1t, 128, t128);
    transpose_w_kernel<<<cdivu(t64, 256), 256, 0, stream>>>(mp_upd_w2, updw2t, 64, t64);
  }

  // ---- embeddings ----
  embed_kernel<19><<<cdivu(NVAR, 4), 256, 0, stream>>>(var_feat, var_w1, var_b1, var_w2, var_b2, h_var, NVAR);
  embed_kernel<5><<<cdivu(NCONS, 4), 256, 0, stream>>>(cons_feat, cons_w1, cons_b1, cons_w2, cons_b2, h_cons, NCONS);
  embed_kernel<30><<<cdivu(NCUT, 4), 256, 0, stream>>>(cut_feat, cut_w1, cut_b1, cut_w2, cut_b2, h_cut, NCUT);

  // ---- edge weights ----
  edge_weight_kernel<<<cdivu(EVC, 4), 128, 0, stream>>>(h_var, h_cons, vc_s, vc_d, vc_ef,
                                                        ewvc_w1, ewvc_b1, ewvc_w2, ewvc_b2, vcw, EVC);
  edge_weight_kernel<<<cdivu(EVK, 4), 128, 0, stream>>>(h_var, h_cut, vk_s, vk_d, vk_ef,
                                                        ewvk_w1, ewvk_b1, ewvk_w2, ewvk_b2, vkw, EVK);

  // ---- one message passing direction ----
  auto mp = [&](float* node, int Nn, const float* neigh, int Nng,
                const int* src, const int* dst, const float* ew, long E, int i) {
    // msg = mlp2(neigh)
    gemm_n64_kernel<64, float, 1><<<cdivu(Nng, 128), 128, 0, stream>>>(
        neigh, msgw1t + (size_t)i * DD * 64, mp_msg_b1 + i * DD, bufA, Nng);
    gemm_n64_kernel<64, float, 0><<<cdivu(Nng, 128), 128, 0, stream>>>(
        bufA, msgw2t + (size_t)i * DD * 64, mp_msg_b2 + i * DD, bufB, Nng);
    // weighted scatter-sum + degree
    zero_kernel<<<cdivu((long)Nn * DD, 256), 256, 0, stream>>>(agg, (long)Nn * DD);
    zero_kernel<<<cdivu((long)Nn, 256), 256, 0, stream>>>(deg, (long)Nn);
    scatter_kernel<<<cdivu(E * DD, 256), 256, 0, stream>>>(bufB, src, dst, ew, agg, deg, E);
    // comb = [agg/deg, node] (bf16)
    build_comb_kernel<<<cdivu((long)Nn * DD, 256), 256, 0, stream>>>(agg, deg, node, comb, Nn);
    // gate = sigmoid(comb@gw+gb)  |  upd = mlp2(comb)
    gemm_n64_kernel<128, __bf16, 2><<<cdivu(Nn, 128), 128, 0, stream>>>(
        comb, gatewt + (size_t)i * DD * 128, mp_gate_b + i * DD, bufB, Nn);
    gemm_n64_kernel<128, __bf16, 1><<<cdivu(Nn, 128), 128, 0, stream>>>(
        comb, updw1t + (size_t)i * DD * 128, mp_upd_b1 + i * DD, bufA, Nn);
    gemm_n64_kernel<64, float, 0><<<cdivu(Nn, 128), 128, 0, stream>>>(
        bufA, updw2t + (size_t)i * DD * 64, mp_upd_b2 + i * DD, bufD, Nn);
    // blend + layernorm (in place on node)
    finalize_ln_kernel<<<cdivu((long)Nn * 32, 128), 128, 0, stream>>>(
        bufB, bufD, mp_ln_g + i * DD, mp_ln_b + i * DD, node, Nn);
  };

  for (int l = 0; l < 4; ++l) {
    mp(h_cons, NCONS, h_var, NVAR, vc_s, vc_d, vcw, EVC, l * 4 + 0);  // v2c
    mp(h_var, NVAR, h_cons, NCONS, vc_d, vc_s, vcw, EVC, l * 4 + 1);  // c2v
    mp(h_cut, NCUT, h_var, NVAR, vk_s, vk_d, vkw, EVK, l * 4 + 2);    // v2k
    mp(h_var, NVAR, h_cut, NCUT, vk_d, vk_s, vkw, EVK, l * 4 + 3);    // k2v
  }
  // h_cut == d_out, nothing more to do.
}
